// TGIFBenchmark_31903017075406
// MI455X (gfx1250) — matrix operand order repaired and checked by reading, not compiled, not run on hardware
//
#include <hip/hip_runtime.h>

// ---------------- types ----------------
typedef __bf16 bf16_t;
typedef __attribute__((ext_vector_type(16))) __bf16 v16bf;
typedef __attribute__((ext_vector_type(8)))  __bf16 v8bf;
typedef __attribute__((ext_vector_type(8)))  float  v8f;
typedef __attribute__((ext_vector_type(4)))  unsigned int u32x4;
typedef __attribute__((ext_vector_type(8)))  int  i32x8;

#define KCHUNK 256

__device__ __forceinline__ bf16_t tobf(float x) { return (bf16_t)x; }

// ---------------------------------------------------------------------
// TDM: async 2D tile load (rows x cols bf16, row stride in elements)
// from global memory into LDS at byte offset lds_off. Rows are packed
// contiguously in LDS (row stride = cols). Issued once per wave; caller
// must gate to one wave and s_wait_tensorcnt + barrier before use.
// Descriptor per CDNA5 ISA 8.3/8.4 (group0 128b, group1 256b),
// VADDR2/3 = NULL (<=2D tensor).
// ---------------------------------------------------------------------
__device__ __forceinline__ void tdm_load_2d_bf16(unsigned lds_off,
                                                 const bf16_t* gptr,
                                                 int rows, int cols,
                                                 int stride_elems)
{
    unsigned long long ga = (unsigned long long)gptr;
    u32x4 g0;
    g0[0] = 1u;                                         // count=1, user D#
    g0[1] = lds_off;                                    // lds_addr (bytes)
    g0[2] = (unsigned)ga;                               // global_addr[31:0]
    g0[3] = (unsigned)((ga >> 32) & 0x01FFFFFFu)        // global_addr[56:32]
          | (2u << 30);                                 // type=2 ("image")
    unsigned long long s0 = (unsigned long long)stride_elems;
    i32x8 g1;
    g1[0] = 0x00010000;                                 // data_size=1 (2B)
    g1[1] = 0;                                          // tensor_dim0 lo16 (=0)
    g1[2] = 0x4000;                                     // tensor_dim0=0x40000000
    g1[3] = (int)(0x4000u | ((unsigned)cols << 16));    // tensor_dim1 hi, tile_dim0
    g1[4] = rows;                                       // tile_dim1 (tile_dim2=0)
    g1[5] = (int)(unsigned)s0;                          // dim0_stride[31:0]
    g1[6] = (int)((s0 >> 32) & 0xFFFFu);                // dim0_stride[47:32]
    g1[7] = 0;                                          // dim1_stride hi (unused)
    asm volatile("tensor_load_to_lds %0, %1" :: "s"(g0), "s"(g1) : "memory");
}

// =====================================================================
// WMMA GEMM:  D[M,N] = A[M,K] * B[N,K]^T (+Cadd[M,N]) (+bias[N])
// A,B bf16 row-major; D,Cadd f32 row-major (ld = ldd).
// block = 256 threads = 8 waves; wave -> MT m-tiles x one 16-col n-tile.
// grid = (N/128, M/(16*MT)). M mult of 16*MT, N mult of 128, K mult of 32.
// A strip (MT*16 rows) staged through LDS via TDM in KCHUNK slices.
// =====================================================================
template<int MT, bool HASC, bool HASBIAS>
__global__ __launch_bounds__(256)
void gemm_bf16_wmma(const bf16_t* __restrict__ A, int lda,
                    const bf16_t* __restrict__ B, int ldb,
                    const float* __restrict__ Cadd,
                    const float* __restrict__ bias,
                    float* __restrict__ D, int ldd,
                    int M, int N, int K)
{
    __shared__ bf16_t sA[MT * 16 * KCHUNK];      // 16/32 KB, LDS offset 0
    const int tid  = threadIdx.x;
    const int lane = tid & 31;
    const int wave = tid >> 5;
    const int gm   = blockIdx.y * (16 * MT);     // M strip origin
    const int gn   = (blockIdx.x * 8 + wave) * 16;
    const int mrow = lane & 15;                  // row (A) / col (B) in tile
    const int sel  = lane >> 4;                  // K-half selector

    v8f acc[MT];
    #pragma unroll
    for (int mt = 0; mt < MT; ++mt) acc[mt] = (v8f){};

    const bf16_t* Brow = B + (size_t)(gn + mrow) * ldb + sel * 16;

    for (int k0 = 0; k0 < K; k0 += KCHUNK) {
        int chunk = K - k0; if (chunk > KCHUNK) chunk = KCHUNK;
        __syncthreads();                         // LDS reuse across slices
        if (wave == 0) {                         // one TDM op per workgroup
            tdm_load_2d_bf16(0u, A + (size_t)gm * lda + k0,
                             MT * 16, chunk, lda);
            __builtin_amdgcn_s_wait_tensorcnt(0);
        }
        __syncthreads();
        if (k0 + KCHUNK < K)                     // prefetch next B slice
            __builtin_prefetch(Brow + k0 + KCHUNK, 0, 1);
        for (int kk = 0; kk < chunk; kk += 32) {
            // B fragment: lane (n, sel) holds 16 contiguous K at kk + sel*16
            v8bf b0 = *(const v8bf*)(Brow + k0 + kk);
            v8bf b1 = *(const v8bf*)(Brow + k0 + kk + 8);
            v16bf b = __builtin_shufflevector(b0, b1,
                0,1,2,3,4,5,6,7,8,9,10,11,12,13,14,15);
            #pragma unroll
            for (int mt = 0; mt < MT; ++mt) {
                // A fragment: lane (m, sel): K = sel*16+{0..15}
                const bf16_t* ar = sA + (mt * 16 + mrow) * chunk + kk;
                v8bf a0 = *(const v8bf*)(ar + sel * 16);
                v8bf a1 = *(const v8bf*)(ar + sel * 16 + 8);
                v16bf a = __builtin_shufflevector(a0, a1,
                    0,1,2,3,4,5,6,7,8,9,10,11,12,13,14,15);
                acc[mt] = __builtin_amdgcn_wmma_f32_16x16x32_bf16(
                    false, a, false, b, (short)0, acc[mt], false, false);
            }
        }
    }

    // C/D layout: VGPR r -> M = r + 8*sel ; N = lane&15
    const unsigned col = (unsigned)(gn + mrow);
    float bv = 0.f;
    if constexpr (HASBIAS) bv = bias[col];
    #pragma unroll
    for (int mt = 0; mt < MT; ++mt) {
        #pragma unroll
        for (int r = 0; r < 8; ++r) {
            unsigned row = (unsigned)(gm + mt * 16 + r + sel * 8);
            unsigned idx = row * (unsigned)ldd + col;
            float v = acc[mt][r] + bv;
            if constexpr (HASC) v += Cadd[idx];
            D[idx] = v;
        }
    }
}

// =====================================================================
// prep / cast kernels
// =====================================================================
__global__ void k_zero(unsigned int* __restrict__ p, long long n)
{
    long long i = (long long)blockIdx.x * 256 + threadIdx.x;
    if (i < n) p[i] = 0u;
}

__global__ void k_cast_concat(const float* __restrict__ s1, int w1,
                              const float* __restrict__ s2, int w2,
                              bf16_t* __restrict__ dst, int wOut, long long total)
{
    long long idx = (long long)blockIdx.x * 256 + threadIdx.x;
    if (idx >= total) return;
    int r = (int)(idx / wOut), c = (int)(idx % wOut);
    float v = 0.f;
    if (c < w1)            v = s1[(size_t)r * w1 + c];
    else if (c - w1 < w2)  v = s2[(size_t)r * w2 + (c - w1)];
    dst[idx] = tobf(v);
}

__global__ void k_castT(const float* __restrict__ src, int rows, int cols,
                        bf16_t* __restrict__ dst, long long total)
{
    long long idx = (long long)blockIdx.x * 256 + threadIdx.x;
    if (idx >= total) return;
    int r = (int)(idx / cols), c = (int)(idx % cols);
    dst[(size_t)c * rows + r] = tobf(src[idx]);
}

__global__ void k_build_feat(const float* __restrict__ vgg,
                             const float* __restrict__ c3d,
                             bf16_t* __restrict__ dst)
{
    long long idx = (long long)blockIdx.x * 256 + threadIdx.x; // < 2048*4096
    int row = (int)(idx >> 12), j = (int)(idx & 4095);
    int t = row >> 5, b = row & 31;
    float v = (j < 2048) ? vgg[(size_t)(b * 64 + t) * 2048 + j]
                         : c3d[(size_t)(b * 64 + t) * 2048 + (j - 2048)];
    dst[idx] = tobf(v);
}

__global__ void k_build_qemb(const int* __restrict__ ques,
                             const float* __restrict__ embW,
                             bf16_t* __restrict__ dst)
{
    long long idx = (long long)blockIdx.x * 256 + threadIdx.x; // < 3200*320
    if (idx >= 3200LL * 320) return;
    int row = (int)(idx / 320), e = (int)(idx % 320);
    int l = row / 160, bc = row % 160;
    float v = 0.f;
    if (e < 300) {
        int tok = ques[bc * 20 + l];
        v = embW[(size_t)tok * 300 + e];
    }
    dst[idx] = tobf(v);
}

__global__ void k_addvec(const float* __restrict__ a, const float* __restrict__ b,
                         float* __restrict__ o, int n)
{
    int i = blockIdx.x * 256 + threadIdx.x;
    if (i < n) o[i] = a[i] + b[i];
}

// =====================================================================
// LSTM pointwise: gates[Nb,2048] (i|f|g|o), c[Nb,512] in/out,
// h -> hcat bf16 (recurrent state), hf f32 (+optional bf16 copy).
// =====================================================================
__global__ void k_lstm_pointwise(const float* __restrict__ gates,
                                 float* __restrict__ c,
                                 bf16_t* __restrict__ hcat, int colOff,
                                 float* __restrict__ hf, int hfStride,
                                 bf16_t* __restrict__ hfb, int hfbStride,
                                 int Nb)
{
    int idx = blockIdx.x * 256 + threadIdx.x;
    if (idx >= Nb * 512) return;
    int r = idx >> 9, j = idx & 511;
    const float* g = gates + (size_t)r * 2048;
    float gi = g[j], gf = g[512 + j], gg = g[1024 + j], go = g[1536 + j];
    float si = 1.f / (1.f + __expf(-gi));
    float sf = 1.f / (1.f + __expf(-gf));
    float so = 1.f / (1.f + __expf(-go));
    float c2 = sf * c[idx] + si * tanhf(gg);
    float h  = so * tanhf(c2);
    c[idx] = c2;
    hcat[(size_t)r * 1024 + colOff + j] = tobf(h);
    if (hf)  hf[(size_t)r * hfStride + j] = h;
    if (hfb) hfb[(size_t)r * hfbStride + j] = tobf(h);
}

__global__ void k_text_init(const bf16_t* __restrict__ hcatV,
                            const float* __restrict__ c1V,
                            const float* __restrict__ c2V,
                            bf16_t* __restrict__ hcatT,
                            float* __restrict__ cT1, float* __restrict__ cT2)
{
    int idx = blockIdx.x * 256 + threadIdx.x;
    if (idx < 160 * 1024) {
        int r = idx >> 10, j = idx & 1023;
        hcatT[idx] = hcatV[(r / 5) * 1024 + j];
    }
    if (idx < 160 * 512) {
        int r = idx >> 9, j = idx & 511;
        cT1[idx] = c1V[(r / 5) * 512 + j];
        cT2[idx] = c2V[(r / 5) * 512 + j];
    }
}

// =====================================================================
// attention + decoder tail
// =====================================================================
__global__ __launch_bounds__(256)
void k_att_score(const float* __restrict__ WsV,   // [B*T,512] (ba folded)
                 const float* __restrict__ Uh,    // [160,512]
                 const float* __restrict__ Va,
                 float* __restrict__ E)           // [160,64]
{
    int wid  = blockIdx.x * 8 + (threadIdx.x >> 5);   // 0..10239
    int lane = threadIdx.x & 31;
    int bc = wid >> 6, t = wid & 63, b = bc / 5;
    const float* ws = WsV + (size_t)(b * 64 + t) * 512;
    const float* uh = Uh  + (size_t)bc * 512;
    float s = 0.f;
    for (int h = lane; h < 512; h += 32)
        s += Va[h] * tanhf(ws[h] + uh[h]);
    for (int off = 16; off > 0; off >>= 1) s += __shfl_xor(s, off);
    if (lane == 0) E[wid] = s;
}

__global__ __launch_bounds__(256)
void k_softmax(const float* __restrict__ E, float* __restrict__ att)
{
    int wid = blockIdx.x * 8 + (threadIdx.x >> 5);
    if (wid >= 160) return;
    int lane = threadIdx.x & 31;
    float v0 = E[wid * 64 + lane], v1 = E[wid * 64 + 32 + lane];
    float m = fmaxf(v0, v1);
    for (int off = 16; off > 0; off >>= 1) m = fmaxf(m, __shfl_xor(m, off));
    float e0 = __expf(v0 - m), e1 = __expf(v1 - m);
    float s = e0 + e1;
    for (int off = 16; off > 0; off >>= 1) s += __shfl_xor(s, off);
    att[wid * 64 + lane]      = e0 / s;
    att[wid * 64 + 32 + lane] = e1 / s;
}

__global__ void k_vid_att(const float* __restrict__ att,
                          const float* __restrict__ sVf,
                          bf16_t* __restrict__ vattbf)
{
    int idx = blockIdx.x * 256 + threadIdx.x;        // < 160*1024
    int bc = idx >> 10, d = idx & 1023, b = bc / 5;
    float s = 0.f;
    for (int t = 0; t < 64; ++t)
        s += att[bc * 64 + t] * sVf[(size_t)(b * 64 + t) * 1024 + d];
    vattbf[idx] = tobf(s);
}

__global__ __launch_bounds__(256)
void k_final(const float* __restrict__ G,
             const float* __restrict__ sTf,
             const float* __restrict__ Wc2, const float* __restrict__ bc2,
             float* __restrict__ out)
{
    int wid = blockIdx.x * 8 + (threadIdx.x >> 5);
    if (wid >= 160) return;
    int lane = threadIdx.x & 31;
    float s = 0.f;
    for (int d = lane; d < 1024; d += 32)
        s += tanhf(G[(size_t)wid * 1024 + d]) * sTf[(size_t)wid * 1024 + d] * Wc2[d];
    for (int off = 16; off > 0; off >>= 1) s += __shfl_xor(s, off);
    if (lane == 0) out[wid] = s + bc2[0];
}

__global__ void k_argmax(const float* __restrict__ out, float* __restrict__ pred)
{
    int b = threadIdx.x;
    if (b >= 32) return;
    float best = out[b * 5]; int bi = 0;
    for (int c = 1; c < 5; ++c) { float v = out[b * 5 + c]; if (v > best) { best = v; bi = c; } }
    pred[b] = (float)bi;
}

// =====================================================================
extern "C" void kernel_launch(void* const* d_in, const int* in_sizes, int n_in,
                              void* d_out, int out_size, void* d_ws, size_t ws_size,
                              hipStream_t stream)
{
    (void)in_sizes; (void)n_in; (void)out_size; (void)ws_size;
    const float* vgg   = (const float*)d_in[0];
    const float* c3d   = (const float*)d_in[1];
    const int*   ques  = (const int*)d_in[2];
    const float* embW  = (const float*)d_in[3];
    const float* t1Wih = (const float*)d_in[4];
    const float* t1Whh = (const float*)d_in[5];
    const float* t1bih = (const float*)d_in[6];
    const float* t1bhh = (const float*)d_in[7];
    const float* t2Wih = (const float*)d_in[8];
    const float* t2Whh = (const float*)d_in[9];
    const float* t2bih = (const float*)d_in[10];
    const float* t2bhh = (const float*)d_in[11];
    const float* v1Wih = (const float*)d_in[12];
    const float* v1Whh = (const float*)d_in[13];
    const float* v1bih = (const float*)d_in[14];
    const float* v1bhh = (const float*)d_in[15];
    const float* v2Wih = (const float*)d_in[16];
    const float* v2Whh = (const float*)d_in[17];
    const float* v2bih = (const float*)d_in[18];
    const float* v2bhh = (const float*)d_in[19];
    const float* Wa    = (const float*)d_in[20];
    const float* Ua    = (const float*)d_in[21];
    const float* Va    = (const float*)d_in[22];
    const float* ba    = (const float*)d_in[23];
    const float* Wc1   = (const float*)d_in[24];
    const float* bc1   = (const float*)d_in[25];
    const float* Wc2   = (const float*)d_in[26];
    const float* bc2   = (const float*)d_in[27];

    char* base = (char*)d_ws; size_t off = 0;
    auto alloc = [&](size_t bytes) -> void* {
        void* p = base + off;
        off += (bytes + 255) & ~(size_t)255;
        return p;
    };
    bf16_t* featbf = (bf16_t*)alloc(2048ull * 4096 * 2);
    bf16_t* Wv1i   = (bf16_t*)alloc(2048ull * 4096 * 2);
    bf16_t* Wv1h   = (bf16_t*)alloc(2048ull * 512 * 2);
    bf16_t* Wv2c   = (bf16_t*)alloc(2048ull * 1024 * 2);
    bf16_t* Wt1i   = (bf16_t*)alloc(2048ull * 320 * 2);
    bf16_t* Wt1h   = (bf16_t*)alloc(2048ull * 512 * 2);
    bf16_t* Wt2c   = (bf16_t*)alloc(2048ull * 1024 * 2);
    bf16_t* WaT    = (bf16_t*)alloc(512ull * 1024 * 2);
    bf16_t* UaT    = (bf16_t*)alloc(512ull * 1024 * 2);
    bf16_t* Wc1b   = (bf16_t*)alloc(1024ull * 1024 * 2);
    bf16_t* qembbf = (bf16_t*)alloc(3200ull * 320 * 2);
    float* Xv    = (float*)alloc(2048ull * 2048 * 4);
    float* Xq    = (float*)alloc(3200ull * 2048 * 4);
    float* gates = (float*)alloc(160ull * 2048 * 4);
    bf16_t* hcatV = (bf16_t*)alloc(32ull * 1024 * 2);
    float* c1V   = (float*)alloc(32ull * 512 * 4);
    float* c2V   = (float*)alloc(32ull * 512 * 4);
    float* sVf   = (float*)alloc(2048ull * 1024 * 4);
    bf16_t* sVbf = (bf16_t*)alloc(2048ull * 1024 * 2);
    bf16_t* hcatT = (bf16_t*)alloc(160ull * 1024 * 2);
    float* cT1   = (float*)alloc(160ull * 512 * 4);
    float* cT2   = (float*)alloc(160ull * 512 * 4);
    float* sTf   = (float*)alloc(160ull * 1024 * 4);
    float* WsV   = (float*)alloc(2048ull * 512 * 4);
    float* UhB   = (float*)alloc(160ull * 512 * 4);
    float* Esc   = (float*)alloc(160ull * 64 * 4);
    float* attW  = (float*)alloc(160ull * 64 * 4);
    bf16_t* vatt = (bf16_t*)alloc(160ull * 1024 * 2);
    float* Gbuf  = (float*)alloc(160ull * 1024 * 4);
    float* bv1   = (float*)alloc(2048 * 4);
    float* bv2   = (float*)alloc(2048 * 4);
    float* bt1   = (float*)alloc(2048 * 4);
    float* bt2   = (float*)alloc(2048 * 4);

    // ---- prep: casts / gathers / biases / zero state ----
    k_build_feat<<<32768, 256, 0, stream>>>(vgg, c3d, featbf);
    k_cast_concat<<<32768, 256, 0, stream>>>(v1Wih, 4096, nullptr, 0, Wv1i, 4096, 2048LL*4096);
    k_cast_concat<<<4096,  256, 0, stream>>>(v1Whh, 512,  nullptr, 0, Wv1h, 512,  2048LL*512);
    k_cast_concat<<<8192,  256, 0, stream>>>(v2Wih, 512,  v2Whh, 512, Wv2c, 1024, 2048LL*1024);
    k_cast_concat<<<2560,  256, 0, stream>>>(t1Wih, 300,  nullptr, 0, Wt1i, 320,  2048LL*320);
    k_cast_concat<<<4096,  256, 0, stream>>>(t1Whh, 512,  nullptr, 0, Wt1h, 512,  2048LL*512);
    k_cast_concat<<<8192,  256, 0, stream>>>(t2Wih, 512,  t2Whh, 512, Wt2c, 1024, 2048LL*1024);
    k_castT<<<2048, 256, 0, stream>>>(Wa, 1024, 512, WaT, 1024LL*512);
    k_castT<<<2048, 256, 0, stream>>>(Ua, 1024, 512, UaT, 1024LL*512);
    k_cast_concat<<<4096, 256, 0, stream>>>(Wc1, 1024, nullptr, 0, Wc1b, 1024, 1024LL*1024);
    k_build_qemb<<<4000, 256, 0, stream>>>(ques, embW, qembbf);
    k_addvec<<<8, 256, 0, stream>>>(v1bih, v1bhh, bv1, 2048);
    k_addvec<<<8, 256, 0, stream>>>(v2bih, v2bhh, bv2, 2048);
    k_addvec<<<8, 256, 0, stream>>>(t1bih, t1bhh, bt1, 2048);
    k_addvec<<<8, 256, 0, stream>>>(t2bih, t2bhh, bt2, 2048);
    k_zero<<<64, 256, 0, stream>>>((unsigned int*)hcatV, 16384);
    k_zero<<<64, 256, 0, stream>>>((unsigned int*)c1V,   16384);
    k_zero<<<64, 256, 0, stream>>>((unsigned int*)c2V,   16384);

    // ---- hoisted input projections (big WMMA GEMMs, MT=4) ----
    gemm_bf16_wmma<4, false, true><<<dim3(16, 32), 256, 0, stream>>>(
        featbf, 4096, Wv1i, 4096, nullptr, bv1, Xv, 2048, 2048, 2048, 4096);
    gemm_bf16_wmma<4, false, true><<<dim3(16, 50), 256, 0, stream>>>(
        qembbf, 320, Wt1i, 320, nullptr, bt1, Xq, 2048, 3200, 2048, 320);

    // ---- video 2-layer LSTM scan (T=64), M=32 -> MT=2 ----
    for (int t = 0; t < 64; ++t) {
        gemm_bf16_wmma<2, true, false><<<dim3(16, 1), 256, 0, stream>>>(
            hcatV, 1024, Wv1h, 512, Xv + (size_t)t * 32 * 2048, nullptr,
            gates, 2048, 32, 2048, 512);
        k_lstm_pointwise<<<64, 256, 0, stream>>>(gates, c1V, hcatV, 0,
            sVf + t * 1024, 64 * 1024, sVbf + t * 1024, 64 * 1024, 32);
        gemm_bf16_wmma<2, false, true><<<dim3(16, 1), 256, 0, stream>>>(
            hcatV, 1024, Wv2c, 1024, nullptr, bv2,
            gates, 2048, 32, 2048, 1024);
        k_lstm_pointwise<<<64, 256, 0, stream>>>(gates, c2V, hcatV, 512,
            sVf + t * 1024 + 512, 64 * 1024, sVbf + t * 1024 + 512, 64 * 1024, 32);
    }

    // ---- broadcast video final state to B*C, text scan (L=20), M=160 ----
    k_text_init<<<640, 256, 0, stream>>>(hcatV, c1V, c2V, hcatT, cT1, cT2);
    for (int l = 0; l < 20; ++l) {
        gemm_bf16_wmma<2, true, false><<<dim3(16, 5), 256, 0, stream>>>(
            hcatT, 1024, Wt1h, 512, Xq + (size_t)l * 160 * 2048, nullptr,
            gates, 2048, 160, 2048, 512);
        k_lstm_pointwise<<<320, 256, 0, stream>>>(gates, cT1, hcatT, 0,
            sTf, 1024, nullptr, 0, 160);
        gemm_bf16_wmma<2, false, true><<<dim3(16, 5), 256, 0, stream>>>(
            hcatT, 1024, Wt2c, 1024, nullptr, bt2,
            gates, 2048, 160, 2048, 1024);
        k_lstm_pointwise<<<320, 256, 0, stream>>>(gates, cT2, hcatT, 512,
            sTf + 512, 1024, nullptr, 0, 160);
    }

    // ---- temporal attention ----
    gemm_bf16_wmma<4, false, true><<<dim3(4, 32), 256, 0, stream>>>(
        sVbf, 1024, WaT, 1024, nullptr, ba, WsV, 512, 2048, 512, 1024);
    gemm_bf16_wmma<2, false, false><<<dim3(4, 5), 256, 0, stream>>>(
        hcatT, 1024, UaT, 1024, nullptr, nullptr, UhB, 512, 160, 512, 1024);
    k_att_score<<<1280, 256, 0, stream>>>(WsV, UhB, Va, Esc);
    k_softmax<<<20, 256, 0, stream>>>(Esc, attW);
    k_vid_att<<<640, 256, 0, stream>>>(attW, sVf, vatt);

    // ---- decoder ----
    gemm_bf16_wmma<2, false, true><<<dim3(8, 5), 256, 0, stream>>>(
        vatt, 1024, Wc1b, 1024, nullptr, bc1, Gbuf, 1024, 160, 1024, 1024);
    k_final<<<20, 256, 0, stream>>>(Gbuf, sTf, Wc2, bc2, (float*)d_out);
    k_argmax<<<1, 32, 0, stream>>>((const float*)d_out, (float*)d_out + 160);
}
